// CustomCrossAttention_30820685316738
// MI455X (gfx1250) — matrix operand order repaired and checked by reading, not compiled, not run on hardware
//
#include <hip/hip_runtime.h>
#include <hip/hip_bf16.h>

// ---------------------------------------------------------------------------
// CustomCrossAttention for MI455X (gfx1250, wave32, WMMA).
//
// All heavy math runs through V_WMMA_F32_16X16X32_BF16 (f32 accumulate).
// Fragment layouts follow CDNA5 ISA:
//   A (16x32 bf16): lanes 0-15 & 16-31 both hold rows M=0..15 (M = lane&15);
//     lane<16:  elems 0..7 = K k0..k0+7,   elems 8..15 = K k0+16..k0+23
//     lane>=16: elems 0..7 = K k0+8..k0+15, elems 8..15 = K k0+24..k0+31
//   B (32x16 bf16): column N = lane&15;
//     same K split as A (lane<16: k0..k0+7, k0+16..k0+23; lane>=16: k0+8..k0+15, k0+24..k0+31)
//   C/D (16x16 f32): VGPR r holds row r (lanes 0-15) / row r+8 (lanes 16-31),
//     column N = lane&15.
// ---------------------------------------------------------------------------

typedef __attribute__((ext_vector_type(16))) __bf16 v16bf;
typedef __attribute__((ext_vector_type(8)))  float  v8f;

#define EMBED 1024
#define NHEAD 16
#define HDIM  64
#define SQL   2048
#define SKV   2048
#define BATCH 2

__device__ __forceinline__ unsigned short f2bf(float f) {
  unsigned int u = __float_as_uint(f);
  u += 0x7FFFu + ((u >> 16) & 1u);          // round-to-nearest-even
  return (unsigned short)(u >> 16);
}

union FragB16 {
  v16bf v;
  uint4 q[2];
  unsigned short u[16];
};

// ---------------------------------------------------------------- stage 0 ---
__global__ __launch_bounds__(256)
void cvt_f32_to_bf16(const float* __restrict__ src,
                     unsigned short* __restrict__ dst, int n) {
  for (int i = blockIdx.x * blockDim.x + threadIdx.x; i < n;
       i += gridDim.x * blockDim.x)
    dst[i] = f2bf(src[i]);
}

// ------------------------------------------------------------ GEMM stage ---
// C[M,N] = A[M,K] @ W[N,K]^T   (torch Linear: W is [out,in], y = x @ W.T)
// mode 0: bf16 out, head-split      [B,H,S,D]
// mode 1: bf16 out, head-split+T    [B,H,D,S]   (for V: PV B-operand contiguous)
// mode 2: f32  out, row-major       [M,N]
__global__ __launch_bounds__(256)
void gemm_bf16_xWT(const unsigned short* __restrict__ A,
                   const unsigned short* __restrict__ W,
                   void* __restrict__ C, int M, int N, int K, int mode) {
  const int lane = threadIdx.x & 31;
  const int wid  = threadIdx.x >> 5;
  const int wm   = wid & 3;         // 4 waves along M  -> 128 rows / block
  const int wn   = wid >> 2;        // 2 waves along N  -> 128 cols / block
  const int ln   = lane & 15;
  const int hs   = lane >> 4;

  const int blockM = blockIdx.y * 128;
  const int blockN = blockIdx.x * 128;

  v8f acc[2][4];
#pragma unroll
  for (int mi = 0; mi < 2; ++mi)
#pragma unroll
    for (int ni = 0; ni < 4; ++ni) acc[mi][ni] = {};

  for (int k0 = 0; k0 < K; k0 += 32) {
    FragB16 af[2];
#pragma unroll
    for (int mi = 0; mi < 2; ++mi) {
      int row = blockM + wm * 32 + mi * 16 + ln;
      const unsigned short* p = A + (size_t)row * K + k0 + hs * 8;
      af[mi].q[0] = *reinterpret_cast<const uint4*>(p);
      af[mi].q[1] = *reinterpret_cast<const uint4*>(p + 16);
    }
    FragB16 bfv[4];
#pragma unroll
    for (int ni = 0; ni < 4; ++ni) {
      int col = blockN + wn * 64 + ni * 16 + ln;
      const unsigned short* p = W + (size_t)col * K + k0 + hs * 8;
      bfv[ni].q[0] = *reinterpret_cast<const uint4*>(p);
      bfv[ni].q[1] = *reinterpret_cast<const uint4*>(p + 16);
    }
#pragma unroll
    for (int mi = 0; mi < 2; ++mi)
#pragma unroll
      for (int ni = 0; ni < 4; ++ni)
        acc[mi][ni] = __builtin_amdgcn_wmma_f32_16x16x32_bf16(
            false, af[mi].v, false, bfv[ni].v, (short)0, acc[mi][ni],
            false, false);
  }

#pragma unroll
  for (int mi = 0; mi < 2; ++mi)
#pragma unroll
    for (int ni = 0; ni < 4; ++ni)
#pragma unroll
      for (int r = 0; r < 8; ++r) {
        int m = blockM + wm * 32 + mi * 16 + r + hs * 8;
        int n = blockN + wn * 64 + ni * 16 + ln;
        float val = acc[mi][ni][r];
        if (mode == 2) {
          reinterpret_cast<float*>(C)[(size_t)m * N + n] = val;
        } else {
          int b = m >> 11, s = m & 2047;   // M = BATCH * 2048
          int h = n >> 6,  d = n & 63;
          size_t dst = (mode == 0)
              ? ((size_t)(b * NHEAD + h) * SQL + s) * HDIM + d
              : ((size_t)(b * NHEAD + h) * HDIM + d) * SKV + s;
          reinterpret_cast<unsigned short*>(C)[dst] = f2bf(val);
        }
      }
}

// ------------------------------------------------------- flash attention ---
// grid (SQ/64, B*H), 128 threads (4 waves), each wave owns 16 query rows.
__global__ __launch_bounds__(128)
void flash_attn(const unsigned short* __restrict__ Qh,   // [B,H,S,D]
                const unsigned short* __restrict__ Kh,   // [B,H,S,D]
                const unsigned short* __restrict__ VT,   // [B,H,D,S]
                const float* __restrict__ bias,          // [B,SKV]
                unsigned short* __restrict__ Oa) {       // [B,S,E] bf16
  __shared__ __align__(16) unsigned short plds[4][16][72]; // P staging, padded
  const int lane = threadIdx.x & 31;
  const int wid  = threadIdx.x >> 5;
  const int ln   = lane & 15;
  const int hs   = lane >> 4;
  const int bh   = blockIdx.y;
  const int b    = bh >> 4;
  const int h    = bh & 15;
  const int q0   = blockIdx.x * 64 + wid * 16;

  const unsigned short* Qbase = Qh + (size_t)bh * SQL * HDIM;
  const unsigned short* Kbase = Kh + (size_t)bh * SKV * HDIM;
  const unsigned short* Vbase = VT + (size_t)bh * HDIM * SKV;
  const float* bb = bias + (size_t)b * SKV;

  // Q A-fragments (K = head dim 64 -> two k-steps), loaded once
  FragB16 qf[2];
  {
    const unsigned short* p = Qbase + (size_t)(q0 + ln) * HDIM;
#pragma unroll
    for (int kk = 0; kk < 2; ++kk) {
      qf[kk].q[0] = *reinterpret_cast<const uint4*>(p + kk * 32 + hs * 8);
      qf[kk].q[1] = *reinterpret_cast<const uint4*>(p + kk * 32 + 16 + hs * 8);
    }
  }

  float mrow[8], lrow[8];
  v8f oacc[4];
#pragma unroll
  for (int r = 0; r < 8; ++r) { mrow[r] = -1e30f; lrow[r] = 0.0f; }
#pragma unroll
  for (int dt = 0; dt < 4; ++dt) oacc[dt] = {};

  const float scale = 0.125f;                    // 1/sqrt(64)
  const float LOG2E = 1.44269504088896340736f;

  for (int kv0 = 0; kv0 < SKV; kv0 += 64) {
    // ---- S = Q K^T * scale + bias  (4 n-subtiles of 16 kv columns) ----
    v8f st[4];
#pragma unroll
    for (int nt = 0; nt < 4; ++nt) {
      v8f s = {};
      int col = kv0 + nt * 16 + ln;
      const unsigned short* kp = Kbase + (size_t)col * HDIM + hs * 8;
#pragma unroll
      for (int kk = 0; kk < 2; ++kk) {
        FragB16 kf;
        kf.q[0] = *reinterpret_cast<const uint4*>(kp + kk * 32);
        kf.q[1] = *reinterpret_cast<const uint4*>(kp + kk * 32 + 16);
        s = __builtin_amdgcn_wmma_f32_16x16x32_bf16(
            false, qf[kk].v, false, kf.v, (short)0, s, false, false);
      }
      float bv = bb[col];
#pragma unroll
      for (int r = 0; r < 8; ++r) s[r] = s[r] * scale + bv;
      st[nt] = s;
    }

    // ---- row max over 64 kv (per-lane over nt, then across 16-lane group) --
    float tm[8];
#pragma unroll
    for (int r = 0; r < 8; ++r)
      tm[r] = fmaxf(fmaxf(st[0][r], st[1][r]), fmaxf(st[2][r], st[3][r]));
#pragma unroll
    for (int off = 1; off < 16; off <<= 1)
#pragma unroll
      for (int r = 0; r < 8; ++r)
        tm[r] = fmaxf(tm[r], __shfl_xor(tm[r], off, 32));

    // ---- online softmax rescale ----
    float corr[8];
#pragma unroll
    for (int r = 0; r < 8; ++r) {
      float mnew = fmaxf(mrow[r], tm[r]);
      corr[r] = exp2f((mrow[r] - mnew) * LOG2E);
      mrow[r] = mnew;
      lrow[r] *= corr[r];
    }
#pragma unroll
    for (int dt = 0; dt < 4; ++dt)
#pragma unroll
      for (int r = 0; r < 8; ++r) oacc[dt][r] *= corr[r];

    // ---- P = exp(S - m); stage bf16 P tile in LDS; accumulate row sums ----
    float rs[8];
#pragma unroll
    for (int r = 0; r < 8; ++r) rs[r] = 0.0f;
#pragma unroll
    for (int nt = 0; nt < 4; ++nt)
#pragma unroll
      for (int r = 0; r < 8; ++r) {
        float p = exp2f((st[nt][r] - mrow[r]) * LOG2E);
        rs[r] += p;
        plds[wid][r + hs * 8][nt * 16 + ln] = f2bf(p);
      }
#pragma unroll
    for (int off = 1; off < 16; off <<= 1)
#pragma unroll
      for (int r = 0; r < 8; ++r) rs[r] += __shfl_xor(rs[r], off, 32);
#pragma unroll
    for (int r = 0; r < 8; ++r) lrow[r] += rs[r];

    // ---- O += P @ V   (A = P from LDS in A-layout, B = V^T contiguous) ----
#pragma unroll
    for (int kk = 0; kk < 2; ++kk) {
      FragB16 pa;
      const unsigned short* pp = &plds[wid][ln][kk * 32 + hs * 8];
      pa.q[0] = *reinterpret_cast<const uint4*>(pp);
      pa.q[1] = *reinterpret_cast<const uint4*>(pp + 16);
#pragma unroll
      for (int dt = 0; dt < 4; ++dt) {
        int drow = dt * 16 + ln;
        const unsigned short* vp =
            Vbase + (size_t)drow * SKV + kv0 + kk * 32 + hs * 8;
        FragB16 vf;
        vf.q[0] = *reinterpret_cast<const uint4*>(vp);
        vf.q[1] = *reinterpret_cast<const uint4*>(vp + 16);
        oacc[dt] = __builtin_amdgcn_wmma_f32_16x16x32_bf16(
            false, pa.v, false, vf.v, (short)0, oacc[dt], false, false);
      }
    }
  }

  // ---- normalize + write bf16 [B,S,E] for the output projection GEMM ----
#pragma unroll
  for (int r = 0; r < 8; ++r) {
    float inv = 1.0f / lrow[r];
    int srow = q0 + r + hs * 8;
#pragma unroll
    for (int dt = 0; dt < 4; ++dt) {
      int e = h * HDIM + dt * 16 + ln;
      Oa[((size_t)b * SQL + srow) * EMBED + e] = f2bf(oacc[dt][r] * inv);
    }
  }
}

// ---------------------------------------------------------------------------
extern "C" void kernel_launch(void* const* d_in, const int* in_sizes, int n_in,
                              void* d_out, int out_size, void* d_ws,
                              size_t ws_size, hipStream_t stream) {
  (void)in_sizes; (void)n_in; (void)out_size; (void)ws_size;
  const float* q  = (const float*)d_in[0];
  const float* k  = (const float*)d_in[1];
  const float* v  = (const float*)d_in[2];
  const float* ab = (const float*)d_in[3];
  const float* Wq = (const float*)d_in[4];
  const float* Wk = (const float*)d_in[5];
  const float* Wv = (const float*)d_in[6];
  const float* Wo = (const float*)d_in[7];

  // workspace layout (bf16 elements): 3 activations + 4 weights + Q/K/V proj
  // + attention output = 32 Mi elements = 64 MB (fits in the 192 MB L2).
  unsigned short* ws = (unsigned short*)d_ws;
  const size_t NX = (size_t)BATCH * SQL * EMBED;   // 4 Mi
  const size_t NW = (size_t)EMBED * EMBED;         // 1 Mi
  unsigned short* XQ  = ws;
  unsigned short* XK  = XQ  + NX;
  unsigned short* XV  = XK  + NX;
  unsigned short* WQb = XV  + NX;
  unsigned short* WKb = WQb + NW;
  unsigned short* WVb = WKb + NW;
  unsigned short* WOb = WVb + NW;
  unsigned short* QH  = WOb + NW;
  unsigned short* KH  = QH  + NX;
  unsigned short* VT  = KH  + NX;
  unsigned short* OA  = VT  + NX;

  cvt_f32_to_bf16<<<2048, 256, 0, stream>>>(q,  XQ,  (int)NX);
  cvt_f32_to_bf16<<<2048, 256, 0, stream>>>(k,  XK,  (int)NX);
  cvt_f32_to_bf16<<<2048, 256, 0, stream>>>(v,  XV,  (int)NX);
  cvt_f32_to_bf16<<<1024, 256, 0, stream>>>(Wq, WQb, (int)NW);
  cvt_f32_to_bf16<<<1024, 256, 0, stream>>>(Wk, WKb, (int)NW);
  cvt_f32_to_bf16<<<1024, 256, 0, stream>>>(Wv, WVb, (int)NW);
  cvt_f32_to_bf16<<<1024, 256, 0, stream>>>(Wo, WOb, (int)NW);

  dim3 gg(EMBED / 128, (BATCH * SQL) / 128);
  gemm_bf16_xWT<<<gg, 256, 0, stream>>>(XQ, WQb, QH, BATCH * SQL, EMBED, EMBED, 0);
  gemm_bf16_xWT<<<gg, 256, 0, stream>>>(XK, WKb, KH, BATCH * SQL, EMBED, EMBED, 0);
  gemm_bf16_xWT<<<gg, 256, 0, stream>>>(XV, WVb, VT, BATCH * SQL, EMBED, EMBED, 1);

  flash_attn<<<dim3(SQL / 64, BATCH * NHEAD), 128, 0, stream>>>(QH, KH, VT, ab, OA);

  gemm_bf16_xWT<<<gg, 256, 0, stream>>>(OA, WOb, d_out, BATCH * SQL, EMBED, EMBED, 2);
}
